// OffsetAttention_43413529428104
// MI455X (gfx1250) — matrix operand order repaired and checked by reading, not compiled, not run on hardware
//
#include <hip/hip_runtime.h>

#define B_ 16
#define C_ 256
#define N_ 2048
#define DA_ 64

typedef __bf16 bf16;
typedef __attribute__((ext_vector_type(16))) __bf16 v16bf;
typedef __attribute__((ext_vector_type(8)))  __bf16 v8bf;
typedef __attribute__((ext_vector_type(8)))  float  v8f;

// ---------- helpers ----------

__device__ inline bf16 f2bf(float f) { return (bf16)f; }   // native v_cvt bf16

__device__ inline unsigned pack2(float a, float b) {       // v_cvt_pk_bf16_f32
    unsigned short x = __builtin_bit_cast(unsigned short, (bf16)a);
    unsigned short y = __builtin_bit_cast(unsigned short, (bf16)b);
    return (unsigned)x | ((unsigned)y << 16);
}

// A-fragment (16x32 bf16) from row-major [row][k] source, leading dim ld (ld%8==0,
// 16B-aligned base). Two contiguous 8-half runs per lane -> two b128 loads.
__device__ inline v16bf load_a_bf(const bf16* p, int ld) {
    int lane = threadIdx.x & 31;
    const bf16* r = p + (lane & 15) * ld + ((lane >> 4) << 4);
    v8bf lo = *(const v8bf*)r;
    v8bf hi = *(const v8bf*)(r + 8);
    v16bf a;
#pragma unroll
    for (int e = 0; e < 8; ++e) { a[e] = lo[e]; a[e + 8] = hi[e]; }
    return a;
}

// B-fragment from fragment-major packed storage: frag f, lane l -> 16 contiguous bf16.
__device__ inline v16bf load_b_packed(const bf16* base, size_t f) {
    return *(const v16bf*)(base + (f * 32 + (threadIdx.x & 31)) * 16);
}

__device__ inline v8f wmma_bf(v16bf a, v16bf b, v8f c) {
    return __builtin_amdgcn_wmma_f32_16x16x32_bf16(false, a, false, b, (short)0, c, false, false);
}

// ---------- repack kernels: build fragment-major B operands ----------

// W (fp32, row-major [D][C]) -> W^T B-fragments. grid = (D/16)*(C/32) frags, block 32.
// frag f = ot*(C/32) + kk;  B[k][col] = W[ot*16+col][kk*32+k]
__global__ void repack_wt_kernel(const float* __restrict__ W, bf16* __restrict__ out) {
    int f = blockIdx.x;
    int kk = f & 7, ot = f >> 3;                 // C/32 == 8
    int lane = threadIdx.x & 31;
    int col = lane & 15, kbase = (lane >> 4) << 4;
    const float* p = W + (size_t)(ot * 16 + col) * C_ + kk * 32 + kbase;
    float4 f0 = *(const float4*)p;
    float4 f1 = *(const float4*)(p + 4);
    float4 f2 = *(const float4*)(p + 8);
    float4 f3 = *(const float4*)(p + 12);
    v16bf b;
    b[0]  = f2bf(f0.x); b[1]  = f2bf(f0.y); b[2]  = f2bf(f0.z); b[3]  = f2bf(f0.w);
    b[4]  = f2bf(f1.x); b[5]  = f2bf(f1.y); b[6]  = f2bf(f1.z); b[7]  = f2bf(f1.w);
    b[8]  = f2bf(f2.x); b[9]  = f2bf(f2.y); b[10] = f2bf(f2.z); b[11] = f2bf(f2.w);
    b[12] = f2bf(f3.x); b[13] = f2bf(f3.y); b[14] = f2bf(f3.z); b[15] = f2bf(f3.w);
    *(v16bf*)(out + ((size_t)f * 32 + lane) * 16) = b;
}

// k [B][DA][N] bf16 -> fragments f = (b*128 + mt)*2 + kk. grid 512, block 256 (8 waves).
__global__ void repack_k_kernel(const bf16* __restrict__ kin, bf16* __restrict__ kout) {
    int g = blockIdx.x * 8 + (threadIdx.x >> 5);
    int lane = threadIdx.x & 31;
    int kk = g & 1, mt = (g >> 1) & 127, b = g >> 8;
    int col = lane & 15, kbase = (lane >> 4) << 4;
    const bf16* src = kin + (size_t)b * DA_ * N_ + (size_t)(kk * 32 + kbase) * N_ + mt * 16 + col;
    v16bf x;
#pragma unroll
    for (int e = 0; e < 16; ++e) x[e] = src[(size_t)e * N_];
    *(v16bf*)(kout + ((size_t)g * 32 + lane) * 16) = x;
}

// v [B][N][C] bf16 -> fragments f = (b*64 + kv)*16 + ot. grid 2048, block 256.
__global__ void repack_v_kernel(const bf16* __restrict__ vin, bf16* __restrict__ vout) {
    int g = blockIdx.x * 8 + (threadIdx.x >> 5);
    int lane = threadIdx.x & 31;
    int ot = g & 15, kv = (g >> 4) & 63, b = g >> 10;
    int col = lane & 15, kbase = (lane >> 4) << 4;
    const bf16* src = vin + (size_t)b * N_ * C_ + (size_t)(kv * 32 + kbase) * C_ + ot * 16 + col;
    v16bf x;
#pragma unroll
    for (int e = 0; e < 16; ++e) x[e] = src[(size_t)e * C_];
    *(v16bf*)(vout + ((size_t)g * 32 + lane) * 16) = x;
}

// ---------- kernel: q/k/v projections (bf16 out, fp32 accumulate) ----------
// grid (N/64, B), block 256 (8 waves). Workgroup handles 64 positions.
__global__ void proj_kernel(const float* __restrict__ x,
                            const bf16* __restrict__ wqp,   // packed Wq^T frags
                            const bf16* __restrict__ wkp,   // packed Wk^T frags
                            const bf16* __restrict__ wvp,   // packed Wv^T frags
                            bf16* __restrict__ q,   // [B][N][DA]
                            bf16* __restrict__ k,   // [B][DA][N]
                            bf16* __restrict__ v)   // [B][N][C]
{
    __shared__ __attribute__((aligned(16))) bf16 xt[64][C_ + 8];   // x^T tile
    const int b = blockIdx.y, n0 = blockIdx.x * 64;
    const float* xb = x + (size_t)b * C_ * N_;
    for (int i = threadIdx.x; i < 64 * C_ / 4; i += blockDim.x) {
        int c = i >> 4;
        int n4 = (i & 15) << 2;
        float4 xv = *(const float4*)(xb + (size_t)c * N_ + n0 + n4);
        xt[n4 + 0][c] = f2bf(xv.x);
        xt[n4 + 1][c] = f2bf(xv.y);
        xt[n4 + 2][c] = f2bf(xv.z);
        xt[n4 + 3][c] = f2bf(xv.w);
    }
    __syncthreads();

    const int wave = threadIdx.x >> 5, lane = threadIdx.x & 31;
    const int col = lane & 15, row_off = (lane >> 4) << 3;
    const v8f vzero = {0, 0, 0, 0, 0, 0, 0, 0};

    // q and k^T: 64x64 output, 16 subtiles, 2 per wave
#pragma unroll
    for (int s = 0; s < 2; ++s) {
        int st = wave * 2 + s;
        int ns = st >> 2, ds = st & 3;
        v8f accq = vzero, acck = vzero;
        for (int kk = 0; kk < C_ / 32; ++kk) {
            v16bf a  = load_a_bf(&xt[ns * 16][kk * 32], C_ + 8);
            v16bf bq = load_b_packed(wqp, ds * 8 + kk);
            v16bf bk = load_b_packed(wkp, ds * 8 + kk);
            accq = wmma_bf(a, bq, accq);
            acck = wmma_bf(a, bk, acck);
        }
        bf16* qb = q + ((size_t)b * N_ + n0 + ns * 16) * DA_ + ds * 16;
        bf16* kb = k + ((size_t)b * DA_ + ds * 16) * N_ + n0 + ns * 16;
#pragma unroll
        for (int r = 0; r < 8; ++r)
            qb[(r + row_off) * DA_ + col] = f2bf(accq[r]);
        uint4 pk;   // k^T rows are 8 consecutive n -> one 16B packed store
        pk.x = pack2(acck[0], acck[1]);
        pk.y = pack2(acck[2], acck[3]);
        pk.z = pack2(acck[4], acck[5]);
        pk.w = pack2(acck[6], acck[7]);
        *(uint4*)(kb + (size_t)col * N_ + row_off) = pk;
    }

    // v: 64x256 output; wave owns 2 o-subtiles x 4 n-subtiles
#pragma unroll
    for (int oi = 0; oi < 2; ++oi) {
        int o0 = (wave * 2 + oi) * 16;
        v8f acc[4];
        for (int i = 0; i < 4; ++i) acc[i] = vzero;
        for (int kk = 0; kk < C_ / 32; ++kk) {
            v16bf bw = load_b_packed(wvp, (o0 >> 4) * 8 + kk);
#pragma unroll
            for (int ns = 0; ns < 4; ++ns) {
                v16bf a = load_a_bf(&xt[ns * 16][kk * 32], C_ + 8);
                acc[ns] = wmma_bf(a, bw, acc[ns]);
            }
        }
#pragma unroll
        for (int ns = 0; ns < 4; ++ns) {
            bf16* vb = v + ((size_t)b * N_ + n0 + ns * 16) * C_ + o0;
#pragma unroll
            for (int r = 0; r < 8; ++r)
                vb[(r + row_off) * C_ + col] = f2bf(acc[ns][r]);
        }
    }
}

// ---------- kernel: per-key-column softmax stats (online max/sum over n) ----------
__global__ void colstats_kernel(const bf16* __restrict__ q,
                                const bf16* __restrict__ kp,
                                float* __restrict__ colmax,
                                float* __restrict__ colinv)
{
    __shared__ float smax[2][64], ssum[2][64];
    const int b = blockIdx.y, m0 = blockIdx.x * 64;
    const int wave = threadIdx.x >> 5, lane = threadIdx.x & 31;
    const int msub = wave & 3, grp = wave >> 2;
    const bf16* qb = q + (size_t)b * N_ * DA_;
    const v8f vzero = {0, 0, 0, 0, 0, 0, 0, 0};

    size_t fk = ((size_t)b * 128 + (m0 >> 4) + msub) * 2;
    v16bf bk0 = load_b_packed(kp, fk + 0);   // d = 0..31
    v16bf bk1 = load_b_packed(kp, fk + 1);   // d = 32..63

    float runM = -3.0e38f, runS = 0.0f;
    for (int n0 = 0; n0 < N_; n0 += 64) {
        if (n0 + 64 < N_)
            __builtin_prefetch(qb + (size_t)(n0 + 64 + grp * 32) * DA_, 0, 1);
#pragma unroll
        for (int s = 0; s < 2; ++s) {
            int ns = grp * 2 + s;
            v16bf a0 = load_a_bf(qb + (size_t)(n0 + ns * 16) * DA_,      DA_);
            v16bf a1 = load_a_bf(qb + (size_t)(n0 + ns * 16) * DA_ + 32, DA_);
            v8f e = vzero;
            e = wmma_bf(a0, bk0, e);
            e = wmma_bf(a1, bk1, e);
            float ml = e[0];
#pragma unroll
            for (int r = 1; r < 8; ++r) ml = fmaxf(ml, e[r]);
            float Mn = fmaxf(runM, ml);
            float sl = 0.0f;
#pragma unroll
            for (int r = 0; r < 8; ++r) sl += __expf(e[r] - Mn);
            runS = runS * __expf(runM - Mn) + sl;
            runM = Mn;
        }
    }
    float oM = __shfl_xor(runM, 16, 32);
    float oS = __shfl_xor(runS, 16, 32);
    float Mn = fmaxf(runM, oM);
    float S  = runS * __expf(runM - Mn) + oS * __expf(oM - Mn);
    if (lane < 16) { smax[grp][msub * 16 + lane] = Mn; ssum[grp][msub * 16 + lane] = S; }
    __syncthreads();
    if (grp == 0 && lane < 16) {
        int m = msub * 16 + lane;
        float M0 = smax[0][m], S0 = ssum[0][m];
        float M1 = smax[1][m], S1 = ssum[1][m];
        float Mx = fmaxf(M0, M1);
        float St = S0 * __expf(M0 - Mx) + S1 * __expf(M1 - Mx);
        colmax[(size_t)b * N_ + m0 + m] = Mx;
        colinv[(size_t)b * N_ + m0 + m] = 1.0f / St;
    }
}

// ---------- kernel: fused attention + output projection + BN partials ----------
// Row-sums computed by WMMA against an all-ones B fragment (rowL = P @ 1).
__global__ void attn_kernel(const float* __restrict__ x,
                            const bf16* __restrict__ q,
                            const bf16* __restrict__ kp,     // packed k frags
                            const bf16* __restrict__ vp,     // packed v frags
                            const float* __restrict__ colmax,
                            const float* __restrict__ colinv,
                            const bf16* __restrict__ wpp,    // packed Wp^T frags
                            float* __restrict__ h,           // [B][C][N]
                            float* __restrict__ bnpart)      // [512][2][C]
{
    __shared__ __attribute__((aligned(16))) bf16 pt[64][64 + 8]; // p tile, D->A relayout
    __shared__ __attribute__((aligned(16))) bf16 yt[64][C_ + 8]; // (x - fsa)^T tile

    const int b = blockIdx.y, n0 = blockIdx.x * 64;
    const int wave = threadIdx.x >> 5, lane = threadIdx.x & 31;
    const int col = lane & 15, row_off = (lane >> 4) << 3;
    const int emsub = wave & 3, egrp = wave >> 2;
    const bf16* qb = q + (size_t)b * N_ * DA_;
    const v8f vzero = {0, 0, 0, 0, 0, 0, 0, 0};

    v16bf bones;
#pragma unroll
    for (int e = 0; e < 16; ++e) bones[e] = f2bf(1.0f);

    // q A-fragments for all 4 n-subtiles, both K chunks (resident whole kernel)
    v16bf aq[4][2];
#pragma unroll
    for (int ns = 0; ns < 4; ++ns) {
        aq[ns][0] = load_a_bf(qb + (size_t)(n0 + ns * 16) * DA_,      DA_);
        aq[ns][1] = load_a_bf(qb + (size_t)(n0 + ns * 16) * DA_ + 32, DA_);
    }

    v8f F[2][4];    // wave owns o-subtiles wave*2+{0,1}, all 4 n-subtiles
    v8f Fs[4];      // row-sum accumulators (P @ ones), per n-subtile
    for (int i = 0; i < 2; ++i)
        for (int j = 0; j < 4; ++j) F[i][j] = vzero;
    for (int j = 0; j < 4; ++j) Fs[j] = vzero;

    for (int m0 = 0; m0 < N_; m0 += 64) {
        if (m0 + 64 < N_) {  // prefetch next packed key/value fragments
            __builtin_prefetch(kp + ((size_t)b * 128 + (m0 >> 4) + 4 + emsub) * 2 * 512, 0, 1);
            __builtin_prefetch(vp + (((size_t)b * 64 + (m0 >> 5) + 2) * 16 + wave * 2) * 512, 0, 1);
        }
        // --- phase 1: energy -> p for this wave's (egrp, emsub) subtiles ---
        float cm = colmax[(size_t)b * N_ + m0 + emsub * 16 + col];
        float ci = colinv[(size_t)b * N_ + m0 + emsub * 16 + col];
        size_t fk = ((size_t)b * 128 + (m0 >> 4) + emsub) * 2;
        v16bf bk0 = load_b_packed(kp, fk + 0);
        v16bf bk1 = load_b_packed(kp, fk + 1);
#pragma unroll
        for (int s = 0; s < 2; ++s) {
            int ns = egrp * 2 + s;
            v8f e = vzero;
            e = wmma_bf(aq[ns][0], bk0, e);
            e = wmma_bf(aq[ns][1], bk1, e);
#pragma unroll
            for (int r = 0; r < 8; ++r)
                pt[ns * 16 + r + row_off][emsub * 16 + col] = f2bf(__expf(e[r] - cm) * ci);
        }
        __syncthreads();
        // --- phase 2: F += P @ V, Fs += P @ 1 (packed v frags straight from L2) ---
#pragma unroll
        for (int kk = 0; kk < 2; ++kk) {
            size_t fv = ((size_t)b * 64 + (m0 >> 5) + kk) * 16 + wave * 2;
            v16bf bv0 = load_b_packed(vp, fv + 0);
            v16bf bv1 = load_b_packed(vp, fv + 1);
#pragma unroll
            for (int ns = 0; ns < 4; ++ns) {
                v16bf ap = load_a_bf(&pt[ns * 16][kk * 32], 64 + 8);
                F[0][ns] = wmma_bf(ap, bv0, F[0][ns]);
                F[1][ns] = wmma_bf(ap, bv1, F[1][ns]);
                Fs[ns]   = wmma_bf(ap, bones, Fs[ns]);
            }
        }
        __syncthreads();
    }

    // --- y = x - fsa (fsa = F / (1e-9 + rowsum)); rowsum lives in Fs same slot ---
    const float* xb = x + (size_t)b * C_ * N_;
#pragma unroll
    for (int oi = 0; oi < 2; ++oi) {
        int o0 = (wave * 2 + oi) * 16;
#pragma unroll
        for (int ns = 0; ns < 4; ++ns) {
            const float* xp = xb + (size_t)(o0 + col) * N_ + n0 + ns * 16 + row_off;
            float4 x0 = *(const float4*)xp;
            float4 x1 = *(const float4*)(xp + 4);
            float xs[8] = {x0.x, x0.y, x0.z, x0.w, x1.x, x1.y, x1.z, x1.w};
#pragma unroll
            for (int r = 0; r < 8; ++r) {
                float fsa = F[oi][ns][r] / (1e-9f + Fs[ns][r]);
                yt[ns * 16 + r + row_off][o0 + col] = f2bf(xs[r] - fsa);
            }
        }
    }
    __syncthreads();

    // --- h = Wp @ y (as h^T = y^T @ Wp^T), plus per-channel BN partial sums ---
    float* hb = h + (size_t)b * C_ * N_;
    const int tile = blockIdx.y * (N_ / 64) + blockIdx.x;
    float* bp = bnpart + (size_t)tile * 2 * C_;
#pragma unroll
    for (int oi = 0; oi < 2; ++oi) {
        int o0 = (wave * 2 + oi) * 16;
        v8f acc[4];
        for (int i = 0; i < 4; ++i) acc[i] = vzero;
        for (int kk = 0; kk < C_ / 32; ++kk) {
            v16bf bw = load_b_packed(wpp, (o0 >> 4) * 8 + kk);
#pragma unroll
            for (int ns = 0; ns < 4; ++ns) {
                v16bf a = load_a_bf(&yt[ns * 16][kk * 32], C_ + 8);
                acc[ns] = wmma_bf(a, bw, acc[ns]);
            }
        }
        float csum = 0.0f, csum2 = 0.0f;
#pragma unroll
        for (int ns = 0; ns < 4; ++ns) {
            float* hp = hb + (size_t)(o0 + col) * N_ + n0 + ns * 16 + row_off;
            *(float4*)hp       = make_float4(acc[ns][0], acc[ns][1], acc[ns][2], acc[ns][3]);
            *(float4*)(hp + 4) = make_float4(acc[ns][4], acc[ns][5], acc[ns][6], acc[ns][7]);
#pragma unroll
            for (int r = 0; r < 8; ++r) {
                float hv = acc[ns][r];
                csum += hv;
                csum2 += hv * hv;
            }
        }
        csum  += __shfl_xor(csum, 16, 32);
        csum2 += __shfl_xor(csum2, 16, 32);
        if (lane < 16) { bp[o0 + lane] = csum; bp[C_ + o0 + lane] = csum2; }
    }
}

// ---------- BN finalize (deterministic two-stage reduction) ----------
__global__ void bnfinal_kernel(const float* __restrict__ bnpart,
                               const float* __restrict__ gamma,
                               const float* __restrict__ beta,
                               float* __restrict__ sca,
                               float* __restrict__ shf)
{
    int o = threadIdx.x;
    if (o >= C_) return;
    float s = 0.0f, s2 = 0.0f;
    for (int t = 0; t < B_ * N_ / 64; ++t) {
        s  += bnpart[(size_t)t * 2 * C_ + o];
        s2 += bnpart[(size_t)t * 2 * C_ + C_ + o];
    }
    const float inv = 1.0f / (float)(B_ * N_);
    float mean = s * inv;
    float var  = s2 * inv - mean * mean;
    float rstd = rsqrtf(var + 1e-5f);
    float g = gamma[o] * rstd;
    sca[o] = g;
    shf[o] = beta[o] - mean * g;
}

// ---------- out = relu(h*sca + shf) + x ----------
__global__ void out_kernel(const float* __restrict__ x,
                           const float* __restrict__ h,
                           const float* __restrict__ sca,
                           const float* __restrict__ shf,
                           float* __restrict__ out)
{
    size_t i = (size_t)blockIdx.x * blockDim.x + threadIdx.x;
    const int per = N_ / 4;
    int o = (int)((i / per) % C_);
    const float4* x4 = (const float4*)x;
    const float4* h4 = (const float4*)h;
    float4* o4 = (float4*)out;
    float a = sca[o], c = shf[o];
    float4 hv = h4[i], xv = x4[i], r;
    r.x = fmaxf(hv.x * a + c, 0.0f) + xv.x;
    r.y = fmaxf(hv.y * a + c, 0.0f) + xv.y;
    r.z = fmaxf(hv.z * a + c, 0.0f) + xv.z;
    r.w = fmaxf(hv.w * a + c, 0.0f) + xv.w;
    o4[i] = r;
}

// ---------- host launch ----------
extern "C" void kernel_launch(void* const* d_in, const int* in_sizes, int n_in,
                              void* d_out, int out_size, void* d_ws, size_t ws_size,
                              hipStream_t stream) {
    (void)in_sizes; (void)n_in; (void)out_size; (void)ws_size;
    const float* x     = (const float*)d_in[0];
    const float* Wq    = (const float*)d_in[1];
    const float* Wk    = (const float*)d_in[2];
    const float* Wv    = (const float*)d_in[3];
    const float* Wp    = (const float*)d_in[4];
    const float* gamma = (const float*)d_in[5];
    const float* beta  = (const float*)d_in[6];

    size_t off = 0;
    char* base = (char*)d_ws;
    auto take = [&](size_t bytes) -> void* {
        void* p = base + off;
        off += (bytes + 255) & ~(size_t)255;
        return p;
    };
    bf16*  q      = (bf16*) take((size_t)B_ * N_ * DA_ * sizeof(bf16));
    bf16*  k      = (bf16*) take((size_t)B_ * DA_ * N_ * sizeof(bf16));
    bf16*  v      = (bf16*) take((size_t)B_ * N_ * C_  * sizeof(bf16));
    bf16*  kpk    = (bf16*) take((size_t)B_ * 128 * 2 * 512 * sizeof(bf16));     // packed k frags
    bf16*  vpk    = (bf16*) take((size_t)B_ * 64 * 16 * 512 * sizeof(bf16));     // packed v frags
    bf16*  wqp    = (bf16*) take((size_t)32  * 512 * sizeof(bf16));
    bf16*  wkp    = (bf16*) take((size_t)32  * 512 * sizeof(bf16));
    bf16*  wvp    = (bf16*) take((size_t)128 * 512 * sizeof(bf16));
    bf16*  wpp    = (bf16*) take((size_t)128 * 512 * sizeof(bf16));
    float* colmax = (float*)take((size_t)B_ * N_ * sizeof(float));
    float* colinv = (float*)take((size_t)B_ * N_ * sizeof(float));
    float* h      = (float*)take((size_t)B_ * C_ * N_ * sizeof(float));
    float* bnpart = (float*)take((size_t)(B_ * N_ / 64) * 2 * C_ * sizeof(float));
    float* sca    = (float*)take((size_t)C_ * sizeof(float));
    float* shf    = (float*)take((size_t)C_ * sizeof(float));

    dim3 blk(256);
    dim3 grd(N_ / 64, B_);
    repack_wt_kernel<<<32,  32, 0, stream>>>(Wq, wqp);
    repack_wt_kernel<<<32,  32, 0, stream>>>(Wk, wkp);
    repack_wt_kernel<<<128, 32, 0, stream>>>(Wv, wvp);
    repack_wt_kernel<<<128, 32, 0, stream>>>(Wp, wpp);
    proj_kernel     <<<grd, blk, 0, stream>>>(x, wqp, wkp, wvp, q, k, v);
    repack_k_kernel <<<512,  blk, 0, stream>>>(k, kpk);
    repack_v_kernel <<<2048, blk, 0, stream>>>(v, vpk);
    colstats_kernel <<<grd, blk, 0, stream>>>(q, kpk, colmax, colinv);
    attn_kernel     <<<grd, blk, 0, stream>>>(x, q, kpk, vpk, colmax, colinv, wpp, h, bnpart);
    bnfinal_kernel  <<<1, C_, 0, stream>>>(bnpart, gamma, beta, sca, shf);
    out_kernel      <<<(B_ * C_ * N_ / 4) / 256, blk, 0, stream>>>(x, h, sca, shf, (float*)d_out);
}
